// ConvBottleneckBlock_34746285425325
// MI455X (gfx1250) — matrix operand order repaired and checked by reading, not compiled, not run on hardware
//
#include <hip/hip_runtime.h>
#include <hip/hip_fp16.h>

typedef __attribute__((ext_vector_type(16))) _Float16 v16h;
typedef __attribute__((ext_vector_type(8)))  float    v8f;

#define D_LEN    4096
#define C_CH     32
#define NTHREADS 256
#define NWAVES   8
#define TILES    (D_LEN / 16)   // 256 tiles of 16 positions
#define LN_EPS   1e-5f
#define HOFF     16             // zero halo before/after h row in LDS

// Branchless exact-GELU: 0.5*v*(1+erf(v/sqrt(2))) with Abramowitz-Stegun 7.1.26
// erf approximation (|err| <= 1.5e-7), using hardware rcp/exp2 transcendentals.
__device__ __forceinline__ float gelu_exact(float v) {
    const float z = fabsf(v) * 0.70710678118654752f;
    const float t = __builtin_amdgcn_rcpf(fmaf(0.3275911f, z, 1.0f));
    float p = fmaf(1.061405429f, t, -1.453152027f);
    p = fmaf(p, t,  1.421413741f);
    p = fmaf(p, t, -0.284496736f);
    p = fmaf(p, t,  0.254829592f);
    p = p * t;
    const float e = __builtin_amdgcn_exp2f(z * z * -1.4426950408889634f);
    const float erf_abs = fmaf(-p, e, 1.0f);           // erf(|z|)
    const float erfv = copysignf(erf_abs, v);
    return 0.5f * v * (1.0f + erfv);
}

__global__ __launch_bounds__(NTHREADS)
void conv_bottleneck_kernel(const float* __restrict__ x,
                            const float* __restrict__ gamma,
                            const float* __restrict__ beta,
                            const float* __restrict__ w_down,   // (32,1,3) -> [c*3+k]
                            const float* __restrict__ b_down,   // (32,)
                            const float* __restrict__ w_up,     // (1,32,3) -> [c*3+k]
                            const float* __restrict__ b_up,     // (1,)
                            float* __restrict__ out)
{
    extern __shared__ float lds[];
    float* s_x   = lds;                      // [4096]
    float* s_h   = s_x + D_LEN;              // [4096 + 32] h[t] at s_h[HOFF+t], zero halos
    float* s_u0  = s_h + (D_LEN + 2 * HOFF); // [4098] u_k with zero halo at 0 and 4097
    float* s_u1  = s_u0 + (D_LEN + 2);
    float* s_u2  = s_u1 + (D_LEN + 2);
    float* s_red = s_u2 + (D_LEN + 2);       // [64]

    const int tid   = threadIdx.x;
    const int wave  = tid >> 5;
    const int lane  = tid & 31;
    const int mlane = lane & 15;
    const int hi    = lane >> 4;
    const int row   = blockIdx.x;
    const float* xrow = x + (size_t)row * D_LEN;

    // zero halos (h head/tail, u endpoints)
    if (tid < HOFF) {
        s_h[tid] = 0.f;
        s_h[HOFF + D_LEN + tid] = 0.f;
    }
    if (tid == 0) {
        s_u0[0] = 0.f; s_u0[D_LEN + 1] = 0.f;
        s_u1[0] = 0.f; s_u1[D_LEN + 1] = 0.f;
        s_u2[0] = 0.f; s_u2[D_LEN + 1] = 0.f;
    }

    // ---- Stage 1: load x row to LDS, accumulate sum / sumsq ----
    float sum = 0.f, sumsq = 0.f;
    for (int p = 0; p < 4; ++p) {
        const int idx = (p * NTHREADS + tid) * 4;
        float4 v = *reinterpret_cast<const float4*>(xrow + idx);
        *reinterpret_cast<float4*>(s_x + idx) = v;
        sum   += (v.x + v.y) + (v.z + v.w);
        sumsq += (v.x * v.x + v.y * v.y) + (v.z * v.z + v.w * v.w);
    }
    for (int off = 16; off > 0; off >>= 1) {
        sum   += __shfl_xor(sum,   off, 32);
        sumsq += __shfl_xor(sumsq, off, 32);
    }
    if (lane == 0) { s_red[wave * 2] = sum; s_red[wave * 2 + 1] = sumsq; }

    // ---- Per-lane WMMA constant operands (A matrices, biases) ----
    // f16 A layout (16x32): lanes 0-15 hold K=0..7 (V0..3) and K=16..23 (V4..7);
    // lanes 16-31 hold K=8..15 and K=24..31. Element i of v16h = K index per that map.
    v16h a_dn0 = {}; v16h a_dn1 = {}; v16h a_up = {};
    if (!hi) {
        a_dn0[0] = (_Float16)w_down[mlane * 3 + 0];
        a_dn0[1] = (_Float16)w_down[mlane * 3 + 1];
        a_dn0[2] = (_Float16)w_down[mlane * 3 + 2];
        a_dn1[0] = (_Float16)w_down[(16 + mlane) * 3 + 0];
        a_dn1[1] = (_Float16)w_down[(16 + mlane) * 3 + 1];
        a_dn1[2] = (_Float16)w_down[(16 + mlane) * 3 + 2];
    }
    if (mlane < 3) {
        // up-conv A: A[m,K] = w_up[ch(K), m]; channel permutation chosen so that
        // the down-conv C/D tiles feed the up-conv B operand with no shuffles.
        const int off = hi ? 8 : 0;
        #pragma unroll
        for (int i = 0; i < 16; ++i)
            a_up[i] = (_Float16)w_up[(off + (i < 8 ? i : i + 8)) * 3 + mlane];
    }
    // down-conv bias folded into the WMMA C accumulator (D = A*B + bias):
    // C/D row r on this lane corresponds to channel (hi?8:0)+r (+16 for tile 1).
    v8f cbias0, cbias1;
    #pragma unroll
    for (int r = 0; r < 8; ++r) {
        cbias0[r] = b_down[(hi ? 8 : 0) + r];
        cbias1[r] = b_down[16 + (hi ? 8 : 0) + r];
    }
    const float bup = b_up[0];

    __syncthreads();

    // ---- mean / rstd ----
    float mu, rs;
    {
        float ts = 0.f, tq = 0.f;
        #pragma unroll
        for (int w = 0; w < NWAVES; ++w) { ts += s_red[w * 2]; tq += s_red[w * 2 + 1]; }
        mu = ts * (1.0f / D_LEN);
        const float var = tq * (1.0f / D_LEN) - mu * mu;
        rs = rsqrtf(var + LN_EPS);
    }

    // ---- Stage 2: layernorm -> s_h (aligned float4 stores; halo offset is 16) ----
    for (int p = 0; p < 4; ++p) {
        const int idx = (p * NTHREADS + tid) * 4;
        float4 v = *reinterpret_cast<const float4*>(s_x + idx);
        float4 g = *reinterpret_cast<const float4*>(gamma + idx);
        float4 b = *reinterpret_cast<const float4*>(beta + idx);
        float4 h;
        h.x = (v.x - mu) * rs * g.x + b.x;
        h.y = (v.y - mu) * rs * g.y + b.y;
        h.z = (v.z - mu) * rs * g.z + b.z;
        h.w = (v.w - mu) * rs * g.w + b.w;
        *reinterpret_cast<float4*>(s_h + HOFF + idx) = h;
    }
    __syncthreads();

    // ---- Stage 3: per-wave tile loop: down-conv WMMA -> GELU -> up-conv WMMA ----
    const float hsel = hi ? 0.f : 1.f;   // branchless zeroing of K=8..15, 24..31 rows (lanes 16-31)
    v16h bdn = {};                       // rows k>=3 stay zero across iterations
    const v8f zacc = {};
    for (int tile = wave; tile < TILES; tile += NWAVES) {
        const int t0 = tile * 16;
        const int t  = t0 + mlane;

        // B (32x16 f16): B[k,n] = h[t0+n+k-1], k=0..2 (unconditional halo'd loads)
        const float hm1 = s_h[HOFF + t - 1] * hsel;
        const float h0  = s_h[HOFF + t    ] * hsel;
        const float hp1 = s_h[HOFF + t + 1] * hsel;
        bdn[0] = (_Float16)hm1;
        bdn[1] = (_Float16)h0;
        bdn[2] = (_Float16)hp1;

        // down conv with bias in the C accumulator: channels 0..15 and 16..31
        v8f d0 = __builtin_amdgcn_wmma_f32_16x16x32_f16(false, a_dn0, false, bdn,
                                                        (short)0, cbias0, false, false);
        v8f d1 = __builtin_amdgcn_wmma_f32_16x16x32_f16(false, a_dn1, false, bdn,
                                                        (short)0, cbias1, false, false);

        // exact GELU (branchless), pack straight into the up-conv B operand:
        // elements 0..7 <- tile0 rows, 8..15 <- tile1 rows (channel-permuted A_up).
        v16h bup_m;
        #pragma unroll
        for (int r = 0; r < 8; ++r) {
            bup_m[r]     = (_Float16)gelu_exact(d0[r]);
            bup_m[8 + r] = (_Float16)gelu_exact(d1[r]);
        }

        // up conv: E[m,n] = sum_c w_up[c,m] * g[c, t0+n]  (rows 0..2 = u_m)
        v8f e = __builtin_amdgcn_wmma_f32_16x16x32_f16(false, a_up, false, bup_m,
                                                       (short)0, zacc, false, false);
        if (!hi) {
            s_u0[t + 1] = e[0];
            s_u1[t + 1] = e[1];
            s_u2[t + 1] = e[2];
        }
    }
    __syncthreads();

    // ---- Stage 4: tap combine + residual: out[t] = x[t] + b_up + u0[t-1]+u1[t]+u2[t+1]
    float* orow = out + (size_t)row * D_LEN;
    for (int p = 0; p < 4; ++p) {
        const int idx = (p * NTHREADS + tid) * 4;
        float4 xv = *reinterpret_cast<const float4*>(s_x + idx);
        float4 o;
        o.x = xv.x + bup + s_u0[idx + 0] + s_u1[idx + 1] + s_u2[idx + 2];
        o.y = xv.y + bup + s_u0[idx + 1] + s_u1[idx + 2] + s_u2[idx + 3];
        o.z = xv.z + bup + s_u0[idx + 2] + s_u1[idx + 3] + s_u2[idx + 4];
        o.w = xv.w + bup + s_u0[idx + 3] + s_u1[idx + 4] + s_u2[idx + 5];
        *reinterpret_cast<float4*>(orow + idx) = o;
    }
}

extern "C" void kernel_launch(void* const* d_in, const int* in_sizes, int n_in,
                              void* d_out, int out_size, void* d_ws, size_t ws_size,
                              hipStream_t stream) {
    (void)n_in; (void)d_ws; (void)ws_size;
    const float* x      = (const float*)d_in[0];
    const float* gamma  = (const float*)d_in[1];
    const float* beta   = (const float*)d_in[2];
    const float* w_down = (const float*)d_in[3];
    const float* b_down = (const float*)d_in[4];
    const float* w_up   = (const float*)d_in[5];
    const float* b_up   = (const float*)d_in[6];
    float* out = (float*)d_out;

    const int rows = out_size / D_LEN;   // 2048
    const size_t shmem =
        (size_t)(D_LEN /*s_x*/ + (D_LEN + 2 * HOFF) /*s_h*/ +
                 3 * (D_LEN + 2) /*u*/ + 64 /*red*/) * sizeof(float);  // ~82 KB

    conv_bottleneck_kernel<<<rows, NTHREADS, shmem, stream>>>(
        x, gamma, beta, w_down, b_down, w_up, b_up, out);
}
